// AttentionModule_21973052686356
// MI455X (gfx1250) — compile-verified, matrix operand order checked by reading, not run on hardware
//
#include <hip/hip_runtime.h>

// ---------------------------------------------------------------------------
// MI455X (gfx1250) implementation of the AttentionModule reference.
// Strategy: bf16 WMMA (v_wmma_f32_16x16x32_bf16) for all big GEMMs, f32
// accumulation/statistics, GroupNorm folded into per-batch effective GEMM
// weights, fused masked-softmax + weighted-sum final kernel.
// Workspace use: ~744 MB.
// ---------------------------------------------------------------------------

typedef __attribute__((ext_vector_type(16))) __bf16 v16bf;
typedef __attribute__((ext_vector_type(8)))  __bf16 v8bf;
typedef __attribute__((ext_vector_type(8)))  float  v8f;

#define NB   4
#define NN   2048
#define KG   32
#define MTOT (NN * KG)   // 65536 positions per batch
#define CVT  256         // total channels of the 256-wide stages

// Build a 16x32 bf16 A/B WMMA fragment for this lane.
// Per ISA layout: lane holds K = {q..q+7} (VGPR0-3) and {16+q..16+q+7}
// (VGPR4-7), q = (lane>=16) ? 8 : 0.  p must already point at (row, kc+q).
__device__ __forceinline__ v16bf load_frag(const __bf16* p) {
    union { v16bf v; v8bf h[2]; } u;
    u.h[0] = *(const v8bf*)(p);
    u.h[1] = *(const v8bf*)(p + 16);
    return u.v;
}

__device__ __forceinline__ v8f wmma_bf16(v16bf a, v16bf b, v8f c) {
    return __builtin_amdgcn_wmma_f32_16x16x32_bf16(
        /*neg_a=*/false, a, /*neg_b=*/false, b,
        /*c_mod=*/(short)0, c, /*reuse_a=*/false, /*reuse_b=*/false);
}

// ---------------------------------------------------------------------------
// Small helper kernels
// ---------------------------------------------------------------------------

__global__ void zero_f32_kernel(float* p, int n) {
    int i = blockIdx.x * 256 + threadIdx.x;
    if (i < n) p[i] = 0.0f;
}

__global__ void wcvt_kernel(const float* __restrict__ in, __bf16* __restrict__ out, int n) {
    int i = blockIdx.x * 256 + threadIdx.x;
    if (i < n) out[i] = (__bf16)in[i];
}

// Transpose [b][C][M] f32  ->  [b][M][C] bf16 (channel-last for WMMA B-frags)
template <int C>
__global__ void transpose_cvt_kernel(const float* __restrict__ in, __bf16* __restrict__ out) {
    __shared__ float tile[32][33];
    int b  = blockIdx.z;
    int c0 = blockIdx.y * 32;
    int m0 = blockIdx.x * 32;
    int tx = threadIdx.x;          // 0..31
    int ty = threadIdx.y;          // 0..7
#pragma unroll
    for (int i = 0; i < 4; ++i) {
        int c = ty + i * 8;
        tile[c][tx] = in[((size_t)(b * C + c0 + c)) * MTOT + m0 + tx];
    }
    __syncthreads();
#pragma unroll
    for (int i = 0; i < 4; ++i) {
        int m = ty + i * 8;
        out[((size_t)b * MTOT + m0 + m) * C + c0 + tx] = (__bf16)tile[tx][m];
    }
}

// feat1[b][n][o] = sum_c W_fc[o][c] * feat[b][c][n] + b_fc[o]   (f32, tiny)
__global__ void feat1_kernel(const float* __restrict__ W, const float* __restrict__ bias,
                             const float* __restrict__ feat, float* __restrict__ feat1) {
    int b = blockIdx.y, n = blockIdx.x, o = threadIdx.x;    // 128 threads
    float acc = bias[o];
#pragma unroll 8
    for (int c = 0; c < 128; ++c)
        acc += W[o * 128 + c] * feat[((size_t)(b * 128 + c)) * NN + n];
    feat1[((size_t)(b * NN + n)) * 128 + o] = acc;
}

// x1 channels [0,128): broadcast relu(feat1) over k, accumulate GN1 stats.
__global__ void fill_feat_half_kernel(const float* __restrict__ feat1,
                                      __bf16* __restrict__ x1,
                                      float* __restrict__ stats1) {
    __shared__ float sh[16][2];
    int b = blockIdx.y, n = blockIdx.x, c = threadIdx.x;    // 128 threads
    if (c < 32) { sh[c >> 1][c & 1] = 0.0f; }
    __syncthreads();
    float v = fmaxf(feat1[((size_t)(b * NN + n)) * 128 + c], 0.0f);
    __bf16 vb = (__bf16)v;
    size_t base = ((size_t)b * MTOT + (size_t)n * KG) * CVT + c;
#pragma unroll
    for (int k = 0; k < KG; ++k) x1[base + (size_t)k * CVT] = vb;
    float s = 32.0f * v, q = 32.0f * v * v;
#pragma unroll
    for (int m = 1; m < 8; m <<= 1) {
        s += __shfl_xor(s, m, 8);
        q += __shfl_xor(q, m, 8);
    }
    int g = c >> 3;    // groups 0..15
    if ((c & 7) == 0) { atomicAdd(&sh[g][0], s); atomicAdd(&sh[g][1], q); }
    __syncthreads();
    if (c < 16) {
        atomicAdd(&stats1[(b * 32 + c) * 2 + 0], sh[c][0]);
        atomicAdd(&stats1[(b * 32 + c) * 2 + 1], sh[c][1]);
    }
}

// Fold a GroupNorm (affine) into the following 1x1 conv:
//   W'[b][o][c] = W[o][c] * (rstd_g * gamma_c)   (bf16)
//   b'[b][o]    = b[o] + sum_c W[o][c] * (beta_c - mean_g * rstd_g * gamma_c)
__global__ void fold_gn_kernel(const float* __restrict__ stats,
                               const float* __restrict__ gamma, const float* __restrict__ beta,
                               const float* __restrict__ Wsrc, const float* __restrict__ bsrc,
                               __bf16* __restrict__ Wdst, float* __restrict__ bdst) {
    int b = blockIdx.x, o = threadIdx.x;     // 4 blocks x 256 threads
    const float invcnt = 1.0f / (8.0f * (float)MTOT);
    float acc = bsrc[o];
    for (int c = 0; c < CVT; ++c) {
        int g = c >> 3;
        float sum = stats[(b * 32 + g) * 2 + 0];
        float sq  = stats[(b * 32 + g) * 2 + 1];
        float mean = sum * invcnt;
        float var  = sq * invcnt - mean * mean;
        float rstd = rsqrtf(var + 1e-5f);
        float s  = rstd * gamma[c];
        float tc = beta[c] - mean * s;
        float w  = Wsrc[o * CVT + c];
        acc += w * tc;
        Wdst[((size_t)(b * CVT + o)) * CVT + c] = (__bf16)(w * s);
    }
    bdst[b * CVT + o] = acc;
}

// GN3 is applied elementwise (relu after norm): emit per-(b,c) scale/shift.
__global__ void fold_gn3_kernel(const float* __restrict__ stats,
                                const float* __restrict__ gamma, const float* __restrict__ beta,
                                float* __restrict__ s3, float* __restrict__ t3) {
    int b = blockIdx.x, c = threadIdx.x;
    const float invcnt = 1.0f / (8.0f * (float)MTOT);
    int g = c >> 3;
    float sum = stats[(b * 32 + g) * 2 + 0];
    float sq  = stats[(b * 32 + g) * 2 + 1];
    float mean = sum * invcnt;
    float var  = sq * invcnt - mean * mean;
    float rstd = rsqrtf(var + 1e-5f);
    float s = rstd * gamma[c];
    s3[b * CVT + c] = s;
    t3[b * CVT + c] = beta[c] - mean * s;
}

// ---------------------------------------------------------------------------
// Main WMMA GEMM:  out[b][m][ch_off+o] = act(W_b[o][:] . X[b][m][:] + bias[o])
// Block: 256 threads = 8 waves, 2(o) x 4(m), block tile 64o x 128m.
// Also accumulates per-(b,group) sum / sumsq for the next GroupNorm.
// ---------------------------------------------------------------------------
template <int CIN, bool RELU, bool OUTF32>
__global__ void gemm_wmma_kernel(const __bf16* __restrict__ W, size_t wstride_b,
                                 const float* __restrict__ bias, int bstride_b,
                                 const __bf16* __restrict__ X,
                                 void* __restrict__ out, int out_ch, int ch_off,
                                 float* __restrict__ stats) {
    __shared__ float sh[8][2];
    int b    = blockIdx.z;
    int tid  = threadIdx.x;
    int wid  = tid >> 5, lane = tid & 31;
    int l15  = lane & 15;
    int q    = (lane & 16) ? 8 : 0;
    int wo   = wid >> 2, wm = wid & 3;
    int o_wave = blockIdx.y * 64 + wo * 32;
    size_t m_wave = (size_t)blockIdx.x * 128 + (size_t)wm * 32;

    const __bf16* Wb = W + wstride_b * b;
    const __bf16* Xb = X + (size_t)b * MTOT * CIN;

    if (tid < 16) sh[tid >> 1][tid & 1] = 0.0f;
    __syncthreads();

    v8f acc[2][2] = {};
#pragma unroll
    for (int kc = 0; kc < CIN; kc += 32) {
        v16bf af[2], bf[2];
#pragma unroll
        for (int ot = 0; ot < 2; ++ot)
            af[ot] = load_frag(Wb + (size_t)(o_wave + ot * 16 + l15) * CIN + kc + q);
#pragma unroll
        for (int mt = 0; mt < 2; ++mt)
            bf[mt] = load_frag(Xb + (m_wave + mt * 16 + l15) * CIN + kc + q);
#pragma unroll
        for (int ot = 0; ot < 2; ++ot)
#pragma unroll
            for (int mt = 0; mt < 2; ++mt)
                acc[ot][mt] = wmma_bf16(af[ot], bf[mt], acc[ot][mt]);
    }

#pragma unroll
    for (int ot = 0; ot < 2; ++ot) {
        int ob = o_wave + ot * 16 + q;                   // 8 consecutive o rows
        v8f bv = *(const v8f*)(bias + (size_t)bstride_b * b + ob);
#pragma unroll
        for (int mt = 0; mt < 2; ++mt) {
            v8f v = acc[ot][mt];
            float lsum = 0.0f, lsq = 0.0f;
            v8f res;
#pragma unroll
            for (int r = 0; r < 8; ++r) {
                float x = v[r] + bv[r];
                if (RELU) x = fmaxf(x, 0.0f);
                res[r] = x;
                lsum += x;
                lsq  += x * x;
            }
            size_t m   = m_wave + mt * 16 + l15;
            size_t off = ((size_t)b * MTOT + m) * out_ch + ch_off + ob;
            if constexpr (OUTF32) {
                *(v8f*)((float*)out + off) = res;
            } else {
                v8bf rb;
#pragma unroll
                for (int r = 0; r < 8; ++r) rb[r] = (__bf16)res[r];
                *(v8bf*)((__bf16*)out + off) = rb;
            }
#pragma unroll
            for (int s = 1; s < 16; s <<= 1) {
                lsum += __shfl_xor(lsum, s, 16);
                lsq  += __shfl_xor(lsq,  s, 16);
            }
            if (l15 == 0) {
                int gi = (wo * 32 + ot * 16 + q) >> 3;   // 0..7 within block
                atomicAdd(&sh[gi][0], lsum);
                atomicAdd(&sh[gi][1], lsq);
            }
        }
    }
    __syncthreads();
    if (tid < 8) {
        int g = ((ch_off + blockIdx.y * 64) >> 3) + tid; // absolute group 0..31
        atomicAdd(&stats[(b * 32 + g) * 2 + 0], sh[tid][0]);
        atomicAdd(&stats[(b * 32 + g) * 2 + 1], sh[tid][1]);
    }
}

// ---------------------------------------------------------------------------
// Final fused kernel, one block per (b,n):
//   scores = W2'_b . x2 + b2'  (WMMA, 256o x 32k x 256c)
//   masked softmax over k, then out[b][o][n] = sum_k relu(t*s3+t3) * weight
// ---------------------------------------------------------------------------
__global__ void final_kernel(const __bf16* __restrict__ W2, const float* __restrict__ b2,
                             const __bf16* __restrict__ X2, const float* __restrict__ T,
                             const float* __restrict__ S3, const float* __restrict__ T3,
                             const int* __restrict__ count, float* __restrict__ out) {
    int b = blockIdx.y, n = blockIdx.x;
    int tid = threadIdx.x;
    int wid = tid >> 5, lane = tid & 31;
    int l15 = lane & 15;
    int q   = (lane & 16) ? 8 : 0;
    int o_wave = wid * 32;

    const __bf16* Wb = W2 + (size_t)b * CVT * CVT;
    const __bf16* Xb = X2 + ((size_t)b * MTOT + (size_t)n * KG) * CVT;

    v8f acc[2][2] = {};   // [o-tile][k-tile]
#pragma unroll
    for (int kc = 0; kc < CVT; kc += 32) {
        v16bf af[2], bf[2];
#pragma unroll
        for (int ot = 0; ot < 2; ++ot)
            af[ot] = load_frag(Wb + (size_t)(o_wave + ot * 16 + l15) * CVT + kc + q);
#pragma unroll
        for (int kt = 0; kt < 2; ++kt)
            bf[kt] = load_frag(Xb + (size_t)(kt * 16 + l15) * CVT + kc + q);
#pragma unroll
        for (int ot = 0; ot < 2; ++ot)
#pragma unroll
            for (int kt = 0; kt < 2; ++kt)
                acc[ot][kt] = wmma_bf16(af[ot], bf[kt], acc[ot][kt]);
    }

    int cnt = count[b * NN + n];
    if (cnt < 1) cnt = 1;

#pragma unroll
    for (int ot = 0; ot < 2; ++ot) {
        int ob = o_wave + ot * 16 + q;                   // 8 consecutive o rows
        v8f bv  = *(const v8f*)(b2 + b * CVT + ob);
        v8f s3v = *(const v8f*)(S3 + b * CVT + ob);
        v8f t3v = *(const v8f*)(T3 + b * CVT + ob);

        float sc[2][8];
#pragma unroll
        for (int kt = 0; kt < 2; ++kt) {
            int k = kt * 16 + l15;
            bool msk = (k >= cnt);
#pragma unroll
            for (int r = 0; r < 8; ++r)
                sc[kt][r] = msk ? -1.0e9f : (acc[ot][kt][r] + bv[r]);
        }
        float mx[8];
#pragma unroll
        for (int r = 0; r < 8; ++r) mx[r] = fmaxf(sc[0][r], sc[1][r]);
#pragma unroll
        for (int s = 1; s < 16; s <<= 1)
#pragma unroll
            for (int r = 0; r < 8; ++r) mx[r] = fmaxf(mx[r], __shfl_xor(mx[r], s, 16));

        float e[2][8], ssum[8];
#pragma unroll
        for (int r = 0; r < 8; ++r) {
            e[0][r] = __expf(sc[0][r] - mx[r]);
            e[1][r] = __expf(sc[1][r] - mx[r]);
            ssum[r] = e[0][r] + e[1][r];
        }
#pragma unroll
        for (int s = 1; s < 16; s <<= 1)
#pragma unroll
            for (int r = 0; r < 8; ++r) ssum[r] += __shfl_xor(ssum[r], s, 16);

        float contrib[8] = {};
#pragma unroll
        for (int kt = 0; kt < 2; ++kt) {
            size_t m = (size_t)n * KG + kt * 16 + l15;
            v8f tv = *(const v8f*)(T + ((size_t)b * MTOT + m) * CVT + ob);
#pragma unroll
            for (int r = 0; r < 8; ++r) {
                float g = fmaxf(tv[r] * s3v[r] + t3v[r], 0.0f);
                contrib[r] += g * (e[kt][r] / ssum[r]);
            }
        }
#pragma unroll
        for (int s = 1; s < 16; s <<= 1)
#pragma unroll
            for (int r = 0; r < 8; ++r) contrib[r] += __shfl_xor(contrib[r], s, 16);

        if (l15 == 0) {
#pragma unroll
            for (int r = 0; r < 8; ++r)
                out[((size_t)(b * CVT + ob + r)) * NN + n] = contrib[r];
        }
    }
}

// ---------------------------------------------------------------------------
// Host launcher
// ---------------------------------------------------------------------------
extern "C" void kernel_launch(void* const* d_in, const int* in_sizes, int n_in,
                              void* d_out, int out_size, void* d_ws, size_t ws_size,
                              hipStream_t stream) {
    (void)in_sizes; (void)n_in; (void)out_size; (void)ws_size;

    const float* feat  = (const float*)d_in[0];
    const float* gf    = (const float*)d_in[1];
    const float* gfo   = (const float*)d_in[2];
    const int*   count = (const int*)  d_in[3];
    const float* W_fc  = (const float*)d_in[4];
    const float* b_fc  = (const float*)d_in[5];
    const float* W_gc  = (const float*)d_in[6];
    const float* b_gc  = (const float*)d_in[7];
    const float* g1    = (const float*)d_in[8];
    const float* be1   = (const float*)d_in[9];
    const float* W_w1  = (const float*)d_in[10];
    const float* b_w1  = (const float*)d_in[11];
    const float* g2    = (const float*)d_in[12];
    const float* be2   = (const float*)d_in[13];
    const float* W_w2  = (const float*)d_in[14];
    const float* b_w2  = (const float*)d_in[15];
    const float* W_fo  = (const float*)d_in[16];
    const float* b_fo  = (const float*)d_in[17];
    const float* g3    = (const float*)d_in[18];
    const float* be3   = (const float*)d_in[19];
    float* out = (float*)d_out;

    // Workspace carve-up (~744 MB total)
    char* base = (char*)d_ws;
    size_t off = 0;
    auto alloc = [&](size_t bytes) -> void* {
        void* p = base + off;
        off = (off + bytes + 255) & ~(size_t)255;
        return p;
    };
    __bf16* gfT    = (__bf16*)alloc((size_t)NB * MTOT * 128 * 2);   //  67 MB
    __bf16* gfoT   = (__bf16*)alloc((size_t)NB * MTOT * 256 * 2);   // 134 MB
    __bf16* x1     = (__bf16*)alloc((size_t)NB * MTOT * 256 * 2);   // 134 MB
    __bf16* x2     = (__bf16*)alloc((size_t)NB * MTOT * 256 * 2);   // 134 MB
    float*  tbuf   = (float*) alloc((size_t)NB * MTOT * 256 * 4);   // 268 MB
    float*  feat1  = (float*) alloc((size_t)NB * NN * 128 * 4);     //   4 MB
    __bf16* Wgc_bf = (__bf16*)alloc(128 * 128 * 2);
    __bf16* Wfo_bf = (__bf16*)alloc(256 * 256 * 2);
    __bf16* W1p    = (__bf16*)alloc((size_t)NB * 256 * 256 * 2);
    __bf16* W2p    = (__bf16*)alloc((size_t)NB * 256 * 256 * 2);
    float*  b1p    = (float*) alloc(NB * 256 * 4);
    float*  b2p    = (float*) alloc(NB * 256 * 4);
    float*  s3     = (float*) alloc(NB * 256 * 4);
    float*  t3     = (float*) alloc(NB * 256 * 4);
    float*  stats  = (float*) alloc(3 * NB * 32 * 2 * 4);
    float* stats1 = stats;
    float* stats2 = stats + NB * 32 * 2;
    float* stats3 = stats + 2 * NB * 32 * 2;

    // 0) zero all GN statistics
    zero_f32_kernel<<<3, 256, 0, stream>>>(stats, 3 * NB * 32 * 2);

    // 1) weight conversions + activation transposes (to [b][m][c] bf16)
    wcvt_kernel<<<(128 * 128 + 255) / 256, 256, 0, stream>>>(W_gc, Wgc_bf, 128 * 128);
    wcvt_kernel<<<(256 * 256 + 255) / 256, 256, 0, stream>>>(W_fo, Wfo_bf, 256 * 256);
    transpose_cvt_kernel<128><<<dim3(MTOT / 32, 128 / 32, NB), dim3(32, 8), 0, stream>>>(gf,  gfT);
    transpose_cvt_kernel<256><<<dim3(MTOT / 32, 256 / 32, NB), dim3(32, 8), 0, stream>>>(gfo, gfoT);

    // 2) feat branch: feat1 = W_fc.feat + b_fc; broadcast relu into x1[:,0:128] + GN1 stats
    feat1_kernel<<<dim3(NN, NB), 128, 0, stream>>>(W_fc, b_fc, feat, feat1);
    fill_feat_half_kernel<<<dim3(NN, NB), 128, 0, stream>>>(feat1, x1, stats1);

    // 3) GEMM A: x1[:,128:256] = relu(W_gc . gfT + b_gc) + GN1 stats (groups 16..31)
    gemm_wmma_kernel<128, true, false><<<dim3(MTOT / 128, 128 / 64, NB), 256, 0, stream>>>(
        Wgc_bf, 0, b_gc, 0, gfT, x1, 256, 128, stats1);

    // 4) fold GN1 into W_w1 -> per-batch W1'/b1'
    fold_gn_kernel<<<NB, 256, 0, stream>>>(stats1, g1, be1, W_w1, b_w1, W1p, b1p);

    // 5) GEMM B: x2 = relu(W1'_b . x1 + b1') + GN2 stats
    gemm_wmma_kernel<256, true, false><<<dim3(MTOT / 128, 256 / 64, NB), 256, 0, stream>>>(
        W1p, (size_t)256 * 256, b1p, 256, x1, x2, 256, 0, stats2);

    // 6) fold GN2 into W_w2 -> per-batch W2'/b2'
    fold_gn_kernel<<<NB, 256, 0, stream>>>(stats2, g2, be2, W_w2, b_w2, W2p, b2p);

    // 7) GEMM D: t = W_fo . gfoT + b_fo (f32, pre-norm) + GN3 stats
    gemm_wmma_kernel<256, false, true><<<dim3(MTOT / 128, 256 / 64, NB), 256, 0, stream>>>(
        Wfo_bf, 0, b_fo, 0, gfoT, tbuf, 256, 0, stats3);

    // 8) GN3 -> elementwise scale/shift
    fold_gn3_kernel<<<NB, 256, 0, stream>>>(stats3, g3, be3, s3, t3);

    // 9) fused scores GEMM + masked softmax + gn3/relu weighted-sum
    final_kernel<<<dim3(NN, NB), 256, 0, stream>>>(W2p, b2p, x2, tbuf, s3, t3, count, out);
}
